// DualAttention_58222576665124
// MI455X (gfx1250) — matrix operand order repaired and checked by reading, not compiled, not run on hardware
//
#include <hip/hip_runtime.h>
#include <hip/hip_bf16.h>

typedef __attribute__((ext_vector_type(16))) __bf16 v16bf;
typedef __attribute__((ext_vector_type(8)))  float  v8f;

static constexpr int B_ = 32, T_ = 512, S_ = 2048, E_ = 256, D_ = 256, F_ = 128;
static constexpr float EPSF = 1e-6f;

// Branchless tanh: one v_exp_f32, no EXEC divergence (libm tanhf is branchy).
__device__ inline float fast_tanh(float x) {
  float a = fabsf(x);
  float t = __expf(-2.0f * a);
  float r = (1.0f - t) / (1.0f + t);
  return copysignf(r, x);
}

// ---------------- WMMA operand loaders (CDNA5 wave32 layouts) ----------------
// A-matrix 16x32 bf16: lane = m + 16*half; elems 0..7 -> K = half*8+e,
// elems 8..15 -> K = 16 + half*8 + e   (ISA 7.12.2, 16-bit A 16x32 table)
__device__ inline v16bf a_frag_f32(const float* __restrict__ base, int ld, int lane) {
  const int m = lane & 15, h = lane >> 4;
  const float* p = base + (long)m * ld + h * 8;
  v16bf a;
#pragma unroll
  for (int e = 0; e < 8; ++e) a[e] = (__bf16)p[e];
#pragma unroll
  for (int e = 0; e < 8; ++e) a[8 + e] = (__bf16)p[16 + e];
  return a;
}

__device__ inline v16bf a_frag_bf16(const __bf16* __restrict__ base, int ld, int lane) {
  const int m = lane & 15, h = lane >> 4;
  const __bf16* p = base + (long)m * ld + h * 8;
  v16bf a;
#pragma unroll
  for (int e = 0; e < 8; ++e) a[e] = p[e];
#pragma unroll
  for (int e = 0; e < 8; ++e) a[8 + e] = p[16 + e];
  return a;
}

// B-matrix 32x16 bf16 from a row-major [N x K] source (i.e. B = src^T):
// lane = n + 16*half ; elems 0..7 -> K = half*8 + e, elems 8..15 -> K = 16 + half*8 + e.
__device__ inline v16bf b_frag_rows_bf16(const __bf16* __restrict__ base, int ld, int lane) {
  const int n = lane & 15, h = lane >> 4;
  const __bf16* p = base + (long)n * ld + h * 8;
  v16bf b;
#pragma unroll
  for (int e = 0; e < 8; ++e) b[e] = p[e];
#pragma unroll
  for (int e = 0; e < 8; ++e) b[8 + e] = p[16 + e];
  return b;
}

__device__ inline v16bf b_frag_rows_f32(const float* __restrict__ base, int ld, int lane) {
  const int n = lane & 15, h = lane >> 4;
  const float* p = base + (long)n * ld + h * 8;
  v16bf b;
#pragma unroll
  for (int e = 0; e < 8; ++e) b[e] = (__bf16)p[e];
#pragma unroll
  for (int e = 0; e < 8; ++e) b[8 + e] = (__bf16)p[16 + e];
  return b;
}

__device__ inline v8f wmma_bf16(v16bf a, v16bf b, v8f c) {
  return __builtin_amdgcn_wmma_f32_16x16x32_bf16(false, a, false, b, (short)0, c, false, false);
}

__device__ inline float redmax16(float v) {
#pragma unroll
  for (int o = 8; o >= 1; o >>= 1) v = fmaxf(v, __shfl_xor(v, o, 16));
  return v;
}
__device__ inline float redsum16(float v) {
#pragma unroll
  for (int o = 8; o >= 1; o >>= 1) v += __shfl_xor(v, o, 16);
  return v;
}

// ---------------- Kernels ----------------

// Y[N,M] = tanh(X[N,K] @ W[M,K]^T + bias[M]) stored as bf16. One 16x16 tile per wave.
__global__ void __launch_bounds__(256)
k_gemm_tanh_bf16(const float* __restrict__ X, const float* __restrict__ W,
                 const float* __restrict__ bias, __bf16* __restrict__ Y,
                 int N, int K, int M) {
  const int lane = threadIdx.x & 31;
  const int tile = blockIdx.x * 8 + (threadIdx.x >> 5);
  const int mt = M >> 4;
  const int tn = tile / mt, tm = tile % mt;
  if (tn >= (N >> 4)) return;  // wave-uniform
  const float* xb = X + (long)tn * 16 * K;
  const float* wb = W + (long)tm * 16 * K;
  v8f c = {};
  for (int k0 = 0; k0 < K; k0 += 32) {
    v16bf a = a_frag_f32(xb + k0, K, lane);
    v16bf b = b_frag_rows_f32(wb + k0, K, lane);
    c = wmma_bf16(a, b, c);
  }
  const int col = (tm << 4) + (lane & 15);
  const float bv = bias[col];
  const int rbase = (tn << 4) + ((lane >> 4) << 3);
#pragma unroll
  for (int r = 0; r < 8; ++r) {
    Y[(long)(rbase + r) * M + col] = (__bf16)fast_tanh(c[r] + bv);
  }
}

// LDS-tiled transpose: ehT[b][e][s] = (bf16) enc[b][s][e]. 32x32 tiles, coalesced both sides.
__global__ void __launch_bounds__(256)
k_transpose_eh(const float* __restrict__ enc, __bf16* __restrict__ ehT) {
  __shared__ __bf16 tile[32][33];
  const int et = E_ / 32, st = S_ / 32;
  const int b   = blockIdx.x / (st * et);
  const int rem = blockIdx.x % (st * et);
  const int s0 = (rem / et) * 32;
  const int e0 = (rem % et) * 32;
  const int tx = threadIdx.x & 31;
  const int ty = threadIdx.x >> 5;
#pragma unroll
  for (int r = ty; r < 32; r += 8)
    tile[r][tx] = (__bf16)enc[((long)b * S_ + s0 + r) * E_ + e0 + tx];
  __syncthreads();
#pragma unroll
  for (int r = ty; r < 32; r += 8)
    ehT[((long)b * E_ + e0 + r) * S_ + s0 + tx] = tile[tx][r];
}

// Fused dual-gated attention: per (b, 16 t-rows), flash-style two passes over S.
__global__ void __launch_bounds__(256)
k_attn_gamma(const __bf16* __restrict__ out_hs, const __bf16* __restrict__ out_fds,
             const __bf16* __restrict__ out2, const __bf16* __restrict__ out3,
             float* __restrict__ gamma) {
  __shared__ float sp[8][16][5];
  __shared__ float fm1[16], fm2[16], fr[16];

  const int lane = threadIdx.x & 31;
  const int wave = threadIdx.x >> 5;
  const int b  = blockIdx.x / (T_ / 16);
  const int t0 = (blockIdx.x % (T_ / 16)) * 16;

  // Preload decoder-side A fragments (16 rows x D), D/32 = 8 chunks each.
  const __bf16* o2 = out2 + ((long)b * T_ + t0) * D_;
  const __bf16* o3 = out3 + ((long)b * T_ + t0) * D_;
  v16bf a2[8], a3[8];
#pragma unroll
  for (int j = 0; j < 8; ++j) {
    a2[j] = a_frag_bf16(o2 + j * 32, D_, lane);
    a3[j] = a_frag_bf16(o3 + j * 32, D_, lane);
  }

  const __bf16* hsb = out_hs  + (long)b * S_ * D_;
  const __bf16* fdb = out_fds + (long)b * S_ * D_;
  const int prow = lane & 15, ph = lane >> 4;  // per-lane prefetch row/chunk

  // Pass 1: online per-row stats (max1, sum1, max2, sum2, sum12).
  float m1[8], s1[8], m2[8], s2[8], s12[8];
#pragma unroll
  for (int r = 0; r < 8; ++r) { m1[r] = m2[r] = -3.0e38f; s1[r] = s2[r] = s12[r] = 0.f; }

  for (int st = wave; st < S_ / 16; st += 8) {
    const int s0 = st * 16;
    // Speculative prefetch of next strided s-tile (global_prefetch_b8).
    {
      const long nx = (long)(s0 + 128 + prow) * D_ + ph * 128;
      __builtin_prefetch(hsb + nx, 0, 1);
      __builtin_prefetch(fdb + nx, 0, 1);
    }
    v8f c1 = {}, c2 = {};
#pragma unroll
    for (int j = 0; j < 8; ++j) {
      v16bf b1 = b_frag_rows_bf16(hsb + (long)s0 * D_ + j * 32, D_, lane);
      c1 = wmma_bf16(a2[j], b1, c1);
      v16bf b2 = b_frag_rows_bf16(fdb + (long)s0 * D_ + j * 32, D_, lane);
      c2 = wmma_bf16(a3[j], b2, c2);
    }
#pragma unroll
    for (int r = 0; r < 8; ++r) {
      float nm1 = fmaxf(m1[r], redmax16(c1[r]));
      float nm2 = fmaxf(m2[r], redmax16(c2[r]));
      float e1 = __expf(c1[r] - nm1);
      float e2 = __expf(c2[r] - nm2);
      float ts1  = redsum16(e1);
      float ts2  = redsum16(e2);
      float ts12 = redsum16(e1 * e2);
      float f1 = __expf(m1[r] - nm1);
      float f2 = __expf(m2[r] - nm2);
      s1[r]  = s1[r]  * f1 + ts1;
      s2[r]  = s2[r]  * f2 + ts2;
      s12[r] = s12[r] * f1 * f2 + ts12;
      m1[r] = nm1; m2[r] = nm2;
    }
  }

  if ((lane & 15) == 0) {
#pragma unroll
    for (int r = 0; r < 8; ++r) {
      int row = r + ((lane >> 4) << 3);
      sp[wave][row][0] = m1[r]; sp[wave][row][1] = s1[r];
      sp[wave][row][2] = m2[r]; sp[wave][row][3] = s2[r];
      sp[wave][row][4] = s12[r];
    }
  }
  __syncthreads();
  if (threadIdx.x < 16) {
    const int row = threadIdx.x;
    float M1 = -3.0e38f, M2 = -3.0e38f;
    for (int w = 0; w < 8; ++w) {
      M1 = fmaxf(M1, sp[w][row][0]);
      M2 = fmaxf(M2, sp[w][row][2]);
    }
    float S1 = 0.f, S2 = 0.f, S12 = 0.f;
    for (int w = 0; w < 8; ++w) {
      float f1 = __expf(sp[w][row][0] - M1);
      float f2 = __expf(sp[w][row][2] - M2);
      S1  += sp[w][row][1] * f1;
      S2  += sp[w][row][3] * f2;
      S12 += sp[w][row][4] * f1 * f2;
    }
    // gamma_s = e1*e2 / (eps*(eps+S1)*(eps+S2) + S12)   (faithful eps-augmented gating)
    float P = (EPSF + S1) * (EPSF + S2);
    fm1[row] = M1; fm2[row] = M2;
    fr[row] = 1.0f / (EPSF * P + S12);
  }
  __syncthreads();

  float l_m1[8], l_m2[8], l_r[8];
  {
    const int half = lane >> 4;
#pragma unroll
    for (int r = 0; r < 8; ++r) {
      int row = r + (half << 3);
      l_m1[r] = fm1[row]; l_m2[r] = fm2[row]; l_r[r] = fr[row];
    }
  }

  // Pass 2: recompute scores, emit gamma.
  for (int st = wave; st < S_ / 16; st += 8) {
    const int s0 = st * 16;
    v8f c1 = {}, c2 = {};
#pragma unroll
    for (int j = 0; j < 8; ++j) {
      v16bf b1 = b_frag_rows_bf16(hsb + (long)s0 * D_ + j * 32, D_, lane);
      c1 = wmma_bf16(a2[j], b1, c1);
      v16bf b2 = b_frag_rows_bf16(fdb + (long)s0 * D_ + j * 32, D_, lane);
      c2 = wmma_bf16(a3[j], b2, c2);
    }
    const int scol = s0 + (lane & 15);
    const int half = lane >> 4;
#pragma unroll
    for (int r = 0; r < 8; ++r) {
      int trow = t0 + r + (half << 3);
      float g = __expf(c1[r] - l_m1[r]) * __expf(c2[r] - l_m2[r]) * l_r[r];
      gamma[((long)b * T_ + trow) * S_ + scol] = g;
    }
  }
}

// context[b,t,e] = sum_s gamma[b,t,s] * ehT[b,e,s]  -> concat[:, 0:E]
__global__ void __launch_bounds__(256)
k_context(const float* __restrict__ gamma, const __bf16* __restrict__ ehT,
          float* __restrict__ concat) {
  const int lane = threadIdx.x & 31;
  const int tile = blockIdx.x * 8 + (threadIdx.x >> 5);
  const int et = E_ / 16, tt = T_ / 16;
  const int b   = tile / (tt * et);
  const int rem = tile % (tt * et);
  const int t0 = (rem / et) * 16;
  const int e0 = (rem % et) * 16;
  const float*  gb = gamma + ((long)b * T_ + t0) * S_;
  const __bf16* eb = ehT + ((long)b * E_ + e0) * S_;  // rows = e, contiguous in s
  v8f c = {};
  for (int s0 = 0; s0 < S_; s0 += 32) {
    v16bf a  = a_frag_f32(gb + s0, S_, lane);
    v16bf bb = b_frag_rows_bf16(eb + s0, S_, lane);
    c = wmma_bf16(a, bb, c);
  }
  const int ecol = e0 + (lane & 15);
  const int half = lane >> 4;
#pragma unroll
  for (int r = 0; r < 8; ++r) {
    int trow = t0 + r + (half << 3);
    concat[((long)b * T_ + trow) * (E_ + D_) + ecol] = c[r];
  }
}

// concat[:, E:E+D] = output
__global__ void k_copy_tail(const float* __restrict__ output, float* __restrict__ concat, int n) {
  int i = blockIdx.x * blockDim.x + threadIdx.x;
  if (i < n) {
    int d = i % D_;
    int bt = i / D_;
    concat[(long)bt * (E_ + D_) + E_ + d] = output[i];
  }
}

// ---------------- Launch ----------------
extern "C" void kernel_launch(void* const* d_in, const int* in_sizes, int n_in,
                              void* d_out, int out_size, void* d_ws, size_t ws_size,
                              hipStream_t stream) {
  (void)in_sizes; (void)n_in; (void)out_size; (void)ws_size;
  const float* output = (const float*)d_in[0];   // [B,T,D]
  const float* enc    = (const float*)d_in[1];   // [S,B,E] (flat == eh view)
  const float* inz    = (const float*)d_in[2];   // [B,S,F]
  const float* W1 = (const float*)d_in[3];
  const float* b1 = (const float*)d_in[4];
  const float* W2 = (const float*)d_in[5];
  const float* b2 = (const float*)d_in[6];
  const float* W3 = (const float*)d_in[7];
  const float* b3 = (const float*)d_in[8];
  const float* W4 = (const float*)d_in[9];
  const float* b4 = (const float*)d_in[10];

  float* concat = (float*)d_out;
  float* gamma  = (float*)d_out + (long)B_ * T_ * (E_ + D_);

  // Workspace layout (bf16 intermediates, ~117 MB)
  __bf16* ws = (__bf16*)d_ws;
  const long nSB = (long)S_ * B_;  // 65536 rows
  __bf16* out_hs  = ws;  ws += nSB * D_;
  __bf16* out_fds = ws;  ws += nSB * D_;
  __bf16* out2b   = ws;  ws += (long)B_ * T_ * D_;
  __bf16* out3b   = ws;  ws += (long)B_ * T_ * D_;
  __bf16* ehT     = ws;  // [B][E][S] bf16

  // ehT = transpose(eh) in bf16 (contiguous-s rows for context WMMA B operand)
  k_transpose_eh<<<B_ * (S_ / 32) * (E_ / 32), 256, 0, stream>>>(enc, ehT);

  // out_hs = tanh(enc @ W1^T + b1): [65536,256]x[256,256] -> 65536 tiles / 8 waves
  k_gemm_tanh_bf16<<<8192, 256, 0, stream>>>(enc, W1, b1, out_hs, (int)nSB, E_, D_);
  // out_fds = tanh(input_z @ W3^T + b3): K = 128
  k_gemm_tanh_bf16<<<8192, 256, 0, stream>>>(inz, W3, b3, out_fds, (int)nSB, F_, D_);
  // out2 = tanh(output @ W2^T + b2); out3 = tanh(output @ W4^T + b4)
  k_gemm_tanh_bf16<<<2048, 256, 0, stream>>>(output, W2, b2, out2b, B_ * T_, D_, D_);
  k_gemm_tanh_bf16<<<2048, 256, 0, stream>>>(output, W4, b4, out3b, B_ * T_, D_, D_);

  // Fused dual-gate attention -> gamma
  k_attn_gamma<<<B_ * (T_ / 16), 256, 0, stream>>>(out_hs, out_fds, out2b, out3b, gamma);

  // context = gamma @ eh -> concat[:, :E]
  k_context<<<2048, 256, 0, stream>>>(gamma, ehT, concat);

  // concat[:, E:] = output
  {
    int n = B_ * T_ * D_;
    k_copy_tail<<<(n + 255) / 256, 256, 0, stream>>>(output, concat, n);
  }
}
